// OTCSurv_48086453846478
// MI455X (gfx1250) — matrix operand order repaired and checked by reading, not compiled, not run on hardware
//
#include <hip/hip_runtime.h>
#include <math.h>

#define NPAT 256
#define NV   5
#define NROW (NPAT * NV)     // 1280 tokens
#define DDX  48
#define DRX  48
#define CDIM 128
#define MCTX 96              // 48 dx + 48 rx code slots per visit
#define DTM  256
#define NHD  8
#define DFFN 2048
#define REPD 128
#define NEGV (-1e20f)

typedef __attribute__((ext_vector_type(16))) _Float16 v16h;
typedef __attribute__((ext_vector_type(8)))  float    v8f;

// ---------------- wave32 reduction ----------------
__device__ inline float wave_sum32(float x) {
#pragma unroll
  for (int off = 16; off > 0; off >>= 1) x += __shfl_xor(x, off, 32);
  return x;
}

// ---------------- WMMA f16 GEMM: C = act(A@B + bias) ----------------
// A: row-major [M,K] fp32 (lda), B: row-major [K,N] fp32 (ldb), C fp32 (ldc).
// One wave per 16x64 output strip: 1 A-frag + 4 B-frags + 4 WMMAs per k-step.
// blockDim=128 (4 waves) -> 64x64 output block. M,N must be multiples of 64.
// Fragment k order here: contiguous, k = 16*(lane/16) + element, same for A and B (wave32).
__device__ inline v16h load_a_frag(const float* __restrict__ A, int lda, int lane) {
  int m  = lane & 15;
  int kh = (lane >> 4) << 4;            // lanes 16-31: K offset +16
  const float* row = A + (size_t)m * lda;
  v16h a;
#pragma unroll
  for (int j = 0; j < 8; ++j) {
    int kb = 2 * j + kh;   // K=2j (contiguous)
    a[2 * j]     = (_Float16)row[kb];
    a[2 * j + 1] = (_Float16)row[kb + 1];
  }
  return a;
}

__device__ inline v16h load_b_frag(const float* __restrict__ B, int ldb, int lane) {
  int n  = lane & 15;
  int kh = (lane >> 4) << 4;            // lanes 16-31: K offset +16
  v16h b;
#pragma unroll
  for (int j = 0; j < 8; ++j) {
    int k = kh + 2 * j;
    b[2 * j]     = (_Float16)B[(size_t)k * ldb + n];
    b[2 * j + 1] = (_Float16)B[(size_t)(k + 1) * ldb + n];
  }
  return b;
}

template <int ACT>  // 0 none, 1 relu, 2 sigmoid
__global__ void gemm_wmma(const float* __restrict__ A, const float* __restrict__ B,
                          const float* __restrict__ bias, float* __restrict__ C,
                          int K, int lda, int ldb, int ldc) {
  int wave = threadIdx.x >> 5;
  int lane = threadIdx.x & 31;
  int m0 = blockIdx.y * 64 + wave * 16;
  int n0 = blockIdx.x * 64;
  v8f acc0 = {}, acc1 = {}, acc2 = {}, acc3 = {};
  for (int k0 = 0; k0 < K; k0 += 32) {
    v16h a  = load_a_frag(A + (size_t)m0 * lda + k0, lda, lane);
    const float* Bk = B + (size_t)k0 * ldb + n0;
    v16h b0 = load_b_frag(Bk,      ldb, lane);
    v16h b1 = load_b_frag(Bk + 16, ldb, lane);
    v16h b2 = load_b_frag(Bk + 32, ldb, lane);
    v16h b3 = load_b_frag(Bk + 48, ldb, lane);
    acc0 = __builtin_amdgcn_wmma_f32_16x16x32_f16(false, a, false, b0, (short)0, acc0, false, false);
    acc1 = __builtin_amdgcn_wmma_f32_16x16x32_f16(false, a, false, b1, (short)0, acc1, false, false);
    acc2 = __builtin_amdgcn_wmma_f32_16x16x32_f16(false, a, false, b2, (short)0, acc2, false, false);
    acc3 = __builtin_amdgcn_wmma_f32_16x16x32_f16(false, a, false, b3, (short)0, acc3, false, false);
  }
  int nl = lane & 15;
  int mh = (lane >> 4) << 3;            // lanes 16-31: M offset +8
  v8f accs[4] = {acc0, acc1, acc2, acc3};
#pragma unroll
  for (int t = 0; t < 4; ++t) {
    int n = n0 + t * 16 + nl;
    float bv = bias[n];
#pragma unroll
    for (int r = 0; r < 8; ++r) {
      int m = m0 + mh + r;
      float v = accs[t][r] + bv;
      if (ACT == 1)      v = fmaxf(v, 0.0f);
      else if (ACT == 2) v = 1.0f / (1.0f + __expf(-v));
      C[(size_t)m * ldc + n] = v;
    }
  }
}

// ---------------- GRAM code-level attention (dx: 3 levels) ----------------
// one wave per (n,v,code); lane holds 4 of 128 channels
__global__ void gram_dx_kernel(const int* __restrict__ x1, const int* __restrict__ x2,
                               const int* __restrict__ x3,
                               const float* __restrict__ E1, const float* __restrict__ E2,
                               const float* __restrict__ E3,
                               const float* __restrict__ We, const float* __restrict__ be,
                               const float* __restrict__ Wev, const float* __restrict__ bev,
                               float* __restrict__ ctx, float* __restrict__ evlog) {
  int gw = (int)((blockIdx.x * blockDim.x + threadIdx.x) >> 5);
  if (gw >= NROW * DDX) return;
  int lane = threadIdx.x & 31;
  int d  = gw % DDX;
  int nv = gw / DDX;
  int i1 = x1[gw], i2 = x2[gw], i3 = x3[gw];
  float4 e1 = ((const float4*)(E1 + (size_t)i1 * CDIM))[lane];
  float4 e2 = ((const float4*)(E2 + (size_t)i2 * CDIM))[lane];
  float4 e3 = ((const float4*)(E3 + (size_t)i3 * CDIM))[lane];
  float4 wl = ((const float4*)We)[lane];
  float4 wh = ((const float4*)(We + CDIM))[lane];
  float s3 = wave_sum32(e3.x * wl.x + e3.y * wl.y + e3.z * wl.z + e3.w * wl.w);
  float t1 = wave_sum32(e1.x * wh.x + e1.y * wh.y + e1.z * wh.z + e1.w * wh.w);
  float t2 = wave_sum32(e2.x * wh.x + e2.y * wh.y + e2.z * wh.z + e2.w * wh.w);
  float t3 = wave_sum32(e3.x * wh.x + e3.y * wh.y + e3.z * wh.z + e3.w * wh.w);
  float b0 = be[0];
  float en0 = fmaxf(s3 + t1 + b0, 0.0f);
  float en1 = fmaxf(s3 + t2 + b0, 0.0f);
  float en2 = fmaxf(s3 + t3 + b0, 0.0f);
  if (i1 == 0) { en0 = NEGV; en1 = NEGV; en2 = NEGV; }
  float mx = fmaxf(en0, fmaxf(en1, en2));
  float a0 = __expf(en0 - mx), a1 = __expf(en1 - mx), a2 = __expf(en2 - mx);
  float inv = 1.0f / (a0 + a1 + a2);
  a0 *= inv; a1 *= inv; a2 *= inv;
  float4 cv;
  cv.x = a0 * e1.x + a1 * e2.x + a2 * e3.x;
  cv.y = a0 * e1.y + a1 * e2.y + a2 * e3.y;
  cv.z = a0 * e1.z + a1 * e2.z + a2 * e3.z;
  cv.w = a0 * e1.w + a1 * e2.w + a2 * e3.w;
  float4 wv = ((const float4*)Wev)[lane];
  float evv = wave_sum32(cv.x * wv.x + cv.y * wv.y + cv.z * wv.z + cv.w * wv.w) + bev[0];
  if (i1 == 0) evv = NEGV;
  ((float4*)(ctx + ((size_t)nv * MCTX + d) * CDIM))[lane] = cv;
  if (lane == 0) evlog[(size_t)nv * MCTX + d] = evv;
}

// ---------------- GRAM rx: 4 levels ----------------
__global__ void gram_rx_kernel(const int* __restrict__ x1, const int* __restrict__ x2,
                               const int* __restrict__ x3, const int* __restrict__ x4,
                               const float* __restrict__ E1, const float* __restrict__ E2,
                               const float* __restrict__ E3, const float* __restrict__ E4,
                               const float* __restrict__ We, const float* __restrict__ be,
                               const float* __restrict__ Wev, const float* __restrict__ bev,
                               float* __restrict__ ctx, float* __restrict__ evlog) {
  int gw = (int)((blockIdx.x * blockDim.x + threadIdx.x) >> 5);
  if (gw >= NROW * DRX) return;
  int lane = threadIdx.x & 31;
  int d  = gw % DRX;
  int nv = gw / DRX;
  int i1 = x1[gw], i2 = x2[gw], i3 = x3[gw], i4 = x4[gw];
  float4 e1 = ((const float4*)(E1 + (size_t)i1 * CDIM))[lane];
  float4 e2 = ((const float4*)(E2 + (size_t)i2 * CDIM))[lane];
  float4 e3 = ((const float4*)(E3 + (size_t)i3 * CDIM))[lane];
  float4 e4 = ((const float4*)(E4 + (size_t)i4 * CDIM))[lane];
  float4 wl = ((const float4*)We)[lane];
  float4 wh = ((const float4*)(We + CDIM))[lane];
  float s4 = wave_sum32(e4.x * wl.x + e4.y * wl.y + e4.z * wl.z + e4.w * wl.w);
  float t1 = wave_sum32(e1.x * wh.x + e1.y * wh.y + e1.z * wh.z + e1.w * wh.w);
  float t2 = wave_sum32(e2.x * wh.x + e2.y * wh.y + e2.z * wh.z + e2.w * wh.w);
  float t3 = wave_sum32(e3.x * wh.x + e3.y * wh.y + e3.z * wh.z + e3.w * wh.w);
  float t4 = wave_sum32(e4.x * wh.x + e4.y * wh.y + e4.z * wh.z + e4.w * wh.w);
  float b0 = be[0];
  float en0 = fmaxf(s4 + t1 + b0, 0.0f);
  float en1 = fmaxf(s4 + t2 + b0, 0.0f);
  float en2 = fmaxf(s4 + t3 + b0, 0.0f);
  float en3 = fmaxf(s4 + t4 + b0, 0.0f);
  if (i1 == 0) { en0 = NEGV; en1 = NEGV; en2 = NEGV; en3 = NEGV; }
  float mx = fmaxf(fmaxf(en0, en1), fmaxf(en2, en3));
  float a0 = __expf(en0 - mx), a1 = __expf(en1 - mx);
  float a2 = __expf(en2 - mx), a3 = __expf(en3 - mx);
  float inv = 1.0f / (a0 + a1 + a2 + a3);
  a0 *= inv; a1 *= inv; a2 *= inv; a3 *= inv;
  float4 cv;
  cv.x = a0 * e1.x + a1 * e2.x + a2 * e3.x + a3 * e4.x;
  cv.y = a0 * e1.y + a1 * e2.y + a2 * e3.y + a3 * e4.y;
  cv.z = a0 * e1.z + a1 * e2.z + a2 * e3.z + a3 * e4.z;
  cv.w = a0 * e1.w + a1 * e2.w + a2 * e3.w + a3 * e4.w;
  float4 wv = ((const float4*)Wev)[lane];
  float evv = wave_sum32(cv.x * wv.x + cv.y * wv.y + cv.z * wv.z + cv.w * wv.w) + bev[0];
  if (i1 == 0) evv = NEGV;
  ((float4*)(ctx + ((size_t)nv * MCTX + DDX + d) * CDIM))[lane] = cv;
  if (lane == 0) evlog[(size_t)nv * MCTX + DDX + d] = evv;
}

// ---------------- visit-level softmax pooling over 96 codes ----------------
// writes ccv into X1[:, 64:192]
__global__ void visit_pool_kernel(const float* __restrict__ ctx,
                                  const float* __restrict__ evlog,
                                  float* __restrict__ X1) {
  int nv = blockIdx.x;
  int t  = threadIdx.x;   // 128
  __shared__ float sv[128];
  __shared__ float att[MCTX];
  const float* ev = evlog + (size_t)nv * MCTX;
  float l = (t < MCTX) ? ev[t] : -1e30f;
  sv[t] = l; __syncthreads();
  for (int s = 64; s > 0; s >>= 1) { if (t < s) sv[t] = fmaxf(sv[t], sv[t + s]); __syncthreads(); }
  float mx = sv[0]; __syncthreads();
  float e = (t < MCTX) ? __expf(l - mx) : 0.0f;
  sv[t] = e; __syncthreads();
  for (int s = 64; s > 0; s >>= 1) { if (t < s) sv[t] += sv[t + s]; __syncthreads(); }
  float inv = 1.0f / sv[0];
  if (t < MCTX) att[t] = e * inv;
  __syncthreads();
  float acc = 0.0f;
  const float* cb = ctx + (size_t)nv * MCTX * CDIM + t;
#pragma unroll 4
  for (int m = 0; m < MCTX; ++m) acc += att[m] * cb[(size_t)m * CDIM];
  X1[(size_t)nv * 192 + 64 + t] = acc;
}

// ---------------- sinusoidal PE add ----------------
__global__ void pe_add_kernel(float* __restrict__ x) {
  int idx = blockIdx.x * blockDim.x + threadIdx.x;
  if (idx >= NROW * DTM) return;
  int c = idx & (DTM - 1);
  int v = (idx >> 8) % NV;
  float expo = (float)(c & ~1) / (float)DTM;
  float ang  = (float)v / powf(10000.0f, expo);
  x[idx] += (c & 1) ? cosf(ang) : sinf(ang);
}

// ---------------- LayerNorm(a + b) ----------------
__global__ void ln_add_kernel(const float* __restrict__ a, const float* __restrict__ b,
                              const float* __restrict__ s, const float* __restrict__ bi,
                              float* __restrict__ out) {
  int row = blockIdx.x;
  int t   = threadIdx.x;  // 256
  __shared__ float red[DTM];
  float x = a[(size_t)row * DTM + t] + b[(size_t)row * DTM + t];
  red[t] = x; __syncthreads();
  for (int st = 128; st > 0; st >>= 1) { if (t < st) red[t] += red[t + st]; __syncthreads(); }
  float mean = red[0] * (1.0f / DTM); __syncthreads();
  float dx = x - mean;
  red[t] = dx * dx; __syncthreads();
  for (int st = 128; st > 0; st >>= 1) { if (t < st) red[t] += red[t + st]; __syncthreads(); }
  float var = red[0] * (1.0f / DTM);
  out[(size_t)row * DTM + t] = dx * rsqrtf(var + 1e-5f) * s[t] + bi[t];
}

// ---------------- MHA core: seq=5, one wave per (n, head) ----------------
__global__ void attn_kernel(const float* __restrict__ qkv, float* __restrict__ out) {
  int blk  = blockIdx.x;           // NPAT * NHD
  int n    = blk >> 3;
  int h    = blk & 7;
  int lane = threadIdx.x;          // 32 = head dim
  const float* base = qkv + (size_t)n * NV * (3 * DTM) + h * 32 + lane;
  float q[NV], k[NV], v[NV];
#pragma unroll
  for (int s = 0; s < NV; ++s) {
    q[s] = base[s * (3 * DTM)];
    k[s] = base[s * (3 * DTM) + DTM];
    v[s] = base[s * (3 * DTM) + 2 * DTM];
  }
  const float scale = 0.17677669529663687f;  // 1/sqrt(32)
  float* ob = out + (size_t)n * NV * DTM + h * 32 + lane;
#pragma unroll
  for (int s = 0; s < NV; ++s) {
    float sc[NV];
#pragma unroll
    for (int t = 0; t < NV; ++t) sc[t] = wave_sum32(q[s] * k[t]) * scale;
    float mx = sc[0];
#pragma unroll
    for (int t = 1; t < NV; ++t) mx = fmaxf(mx, sc[t]);
    float sum = 0.0f;
#pragma unroll
    for (int t = 0; t < NV; ++t) { sc[t] = __expf(sc[t] - mx); sum += sc[t]; }
    float inv = 1.0f / sum;
    float acc = 0.0f;
#pragma unroll
    for (int t = 0; t < NV; ++t) acc += sc[t] * v[t];
    ob[s * DTM] = acc * inv;
  }
}

// ---------------- visit compression over V=5 ----------------
__global__ void compress_kernel(const float* __restrict__ enc, const float* __restrict__ Wcv,
                                const float* __restrict__ bcv, float* __restrict__ civ) {
  int n = blockIdx.x;
  int t = threadIdx.x;   // 128
  __shared__ float red[REPD];
  float w = Wcv[t];
  float lg[NV], e[NV];
#pragma unroll
  for (int v = 0; v < NV; ++v) {
    e[v] = enc[((size_t)n * NV + v) * REPD + t];
    red[t] = e[v] * w; __syncthreads();
    for (int s = 64; s > 0; s >>= 1) { if (t < s) red[t] += red[t + s]; __syncthreads(); }
    lg[v] = red[0] + bcv[0];
    __syncthreads();
  }
  float mx = lg[0];
#pragma unroll
  for (int v = 1; v < NV; ++v) mx = fmaxf(mx, lg[v]);
  float sum = 0.0f, a[NV];
#pragma unroll
  for (int v = 0; v < NV; ++v) { a[v] = __expf(lg[v] - mx); sum += a[v]; }
  float inv = 1.0f / sum;
  float acc = 0.0f;
#pragma unroll
  for (int v = 0; v < NV; ++v) acc += a[v] * e[v];
  civ[(size_t)n * REPD + t] = acc * inv;
}

// =====================================================================
extern "C" void kernel_launch(void* const* d_in, const int* in_sizes, int n_in,
                              void* d_out, int out_size, void* d_ws, size_t ws_size,
                              hipStream_t stream) {
  (void)in_sizes; (void)n_in; (void)out_size; (void)ws_size;
  const int*   x_dx_l1 = (const int*)d_in[0];
  const int*   x_dx_l2 = (const int*)d_in[1];
  const int*   x_dx_l3 = (const int*)d_in[2];
  const int*   x_rx_l1 = (const int*)d_in[3];
  const int*   x_rx_l2 = (const int*)d_in[4];
  const int*   x_rx_l3 = (const int*)d_in[5];
  const int*   x_rx_l4 = (const int*)d_in[6];
  const float* x_demo  = (const float*)d_in[7];
  const float* W_demo  = (const float*)d_in[8];
  const float* b_demo  = (const float*)d_in[9];
  const float* E_dx1   = (const float*)d_in[10];
  const float* E_dx2   = (const float*)d_in[11];
  const float* E_dx3   = (const float*)d_in[12];
  const float* E_rx1   = (const float*)d_in[13];
  const float* E_rx2   = (const float*)d_in[14];
  const float* E_rx3   = (const float*)d_in[15];
  const float* E_rx4   = (const float*)d_in[16];
  const float* We_dx   = (const float*)d_in[17];
  const float* be_dx   = (const float*)d_in[18];
  const float* We_rx   = (const float*)d_in[19];
  const float* be_rx   = (const float*)d_in[20];
  const float* Wev_dx  = (const float*)d_in[21];
  const float* bev_dx  = (const float*)d_in[22];
  const float* Wev_rx  = (const float*)d_in[23];
  const float* bev_rx  = (const float*)d_in[24];
  const float* Wcv     = (const float*)d_in[25];
  const float* bcv     = (const float*)d_in[26];
  const float* enc_w1  = (const float*)d_in[27];
  const float* enc_b1  = (const float*)d_in[28];
  const float* qkv_w   = (const float*)d_in[29];
  const float* qkv_b   = (const float*)d_in[30];
  const float* ow      = (const float*)d_in[31];
  const float* ob      = (const float*)d_in[32];
  const float* ln1s    = (const float*)d_in[33];
  const float* ln1b    = (const float*)d_in[34];
  const float* ff1w    = (const float*)d_in[35];
  const float* ff1b    = (const float*)d_in[36];
  const float* ff2w    = (const float*)d_in[37];
  const float* ff2b    = (const float*)d_in[38];
  const float* ln2s    = (const float*)d_in[39];
  const float* ln2b    = (const float*)d_in[40];
  const float* enc_w2  = (const float*)d_in[41];
  const float* enc_b2  = (const float*)d_in[42];
  const float* mlp_w   = (const float*)d_in[43];
  const float* mlp_b   = (const float*)d_in[44];
  const float* mlp2_w  = (const float*)d_in[45];
  const float* mlp2_b  = (const float*)d_in[46];
  const float* fc1w    = (const float*)d_in[47];
  const float* fc1b    = (const float*)d_in[48];
  const float* fc2w    = (const float*)d_in[49];
  const float* fc2b    = (const float*)d_in[50];
  const float* fc3w    = (const float*)d_in[51];
  const float* fc3b    = (const float*)d_in[52];

  // workspace layout (floats)
  float* ws = (float*)d_ws;
  size_t off = 0;
  float* ctx   = ws + off; off += (size_t)NROW * MCTX * CDIM;   // 15.7M
  float* evlog = ws + off; off += (size_t)NROW * MCTX;
  float* X1    = ws + off; off += (size_t)NROW * 192;
  float* xb    = ws + off; off += (size_t)NROW * DTM;
  float* qkvb  = ws + off; off += (size_t)NROW * 3 * DTM;
  float* attb  = ws + off; off += (size_t)NROW * DTM;
  float* projb = ws + off; off += (size_t)NROW * DTM;
  float* ffb   = ws + off; off += (size_t)NROW * DFFN;
  float* ff2o  = ws + off; off += (size_t)NROW * DTM;
  float* encb  = ws + off; off += (size_t)NROW * REPD;
  float* t1b   = ws + off; off += (size_t)NPAT * 64;
  float* hb    = ws + off; off += (size_t)NPAT * 256;
  float* h2b   = ws + off; off += (size_t)NPAT * 128;

  float* outp = (float*)d_out;                 // [256,2,64]
  float* civ  = outp + (size_t)NPAT * 2 * 64;  // [256,128]
  float* ph   = civ  + (size_t)NPAT * REPD;    // [256,64]

  // ---- GRAM code-level context ----
  {
    int waves  = NROW * DDX;            // 61440
    int blocks = waves / 8;             // 256 thr = 8 waves
    gram_dx_kernel<<<blocks, 256, 0, stream>>>(x_dx_l1, x_dx_l2, x_dx_l3,
        E_dx1, E_dx2, E_dx3, We_dx, be_dx, Wev_dx, bev_dx, ctx, evlog);
    gram_rx_kernel<<<blocks, 256, 0, stream>>>(x_rx_l1, x_rx_l2, x_rx_l3, x_rx_l4,
        E_rx1, E_rx2, E_rx3, E_rx4, We_rx, be_rx, Wev_rx, bev_rx, ctx, evlog);
  }
  // ---- visit pooling -> X1[:,64:192] ; demo -> X1[:,0:64] ----
  visit_pool_kernel<<<NROW, 128, 0, stream>>>(ctx, evlog, X1);
  gemm_wmma<1><<<dim3(1, 20), 128, 0, stream>>>(x_demo, W_demo, b_demo, X1,
      32, 32, 64, 192);
  // ---- enc1 + PE ----
  gemm_wmma<0><<<dim3(4, 20), 128, 0, stream>>>(X1, enc_w1, enc_b1, xb,
      192, 192, DTM, DTM);
  pe_add_kernel<<<(NROW * DTM + 255) / 256, 256, 0, stream>>>(xb);
  // ---- 4 transformer layers ----
  for (int i = 0; i < 4; ++i) {
    gemm_wmma<0><<<dim3(12, 20), 128, 0, stream>>>(xb, qkv_w + (size_t)i * DTM * 3 * DTM,
        qkv_b + (size_t)i * 3 * DTM, qkvb, DTM, DTM, 3 * DTM, 3 * DTM);
    attn_kernel<<<NPAT * NHD, 32, 0, stream>>>(qkvb, attb);
    gemm_wmma<0><<<dim3(4, 20), 128, 0, stream>>>(attb, ow + (size_t)i * DTM * DTM,
        ob + (size_t)i * DTM, projb, DTM, DTM, DTM, DTM);
    ln_add_kernel<<<NROW, DTM, 0, stream>>>(xb, projb, ln1s + (size_t)i * DTM,
        ln1b + (size_t)i * DTM, xb);
    gemm_wmma<1><<<dim3(32, 20), 128, 0, stream>>>(xb, ff1w + (size_t)i * DTM * DFFN,
        ff1b + (size_t)i * DFFN, ffb, DTM, DTM, DFFN, DFFN);
    gemm_wmma<0><<<dim3(4, 20), 128, 0, stream>>>(ffb, ff2w + (size_t)i * DFFN * DTM,
        ff2b + (size_t)i * DTM, ff2o, DFFN, DFFN, DTM, DTM);
    ln_add_kernel<<<NROW, DTM, 0, stream>>>(xb, ff2o, ln2s + (size_t)i * DTM,
        ln2b + (size_t)i * DTM, xb);
  }
  // ---- enc2 + visit compression ----
  gemm_wmma<0><<<dim3(2, 20), 128, 0, stream>>>(xb, enc_w2, enc_b2, encb,
      DTM, DTM, REPD, REPD);
  compress_kernel<<<NPAT, REPD, 0, stream>>>(encb, Wcv, bcv, civ);
  // ---- ph head ----
  gemm_wmma<1><<<dim3(1, 4), 128, 0, stream>>>(civ, mlp_w, mlp_b, t1b,
      REPD, REPD, 64, 64);
  gemm_wmma<0><<<dim3(1, 4), 128, 0, stream>>>(t1b, mlp2_w, mlp2_b, ph,
      64, 64, 64, 64);
  // ---- per-event FC heads ----
  for (int e = 0; e < 2; ++e) {
    gemm_wmma<1><<<dim3(4, 4), 128, 0, stream>>>(civ, fc1w + (size_t)e * REPD * 256,
        fc1b + (size_t)e * 256, hb, REPD, REPD, 256, 256);
    gemm_wmma<1><<<dim3(2, 4), 128, 0, stream>>>(hb, fc2w + (size_t)e * 256 * 128,
        fc2b + (size_t)e * 128, h2b, 256, 256, 128, 128);
    gemm_wmma<2><<<dim3(1, 4), 128, 0, stream>>>(h2b, fc3w + (size_t)e * 128 * 64,
        fc3b + (size_t)e * 64, outp + e * 64, 128, 128, 64, 128);
  }
}
